// NLBlock_76982993813824
// MI455X (gfx1250) — matrix operand order repaired and checked by reading, not compiled, not run on hardware
//
#include <hip/hip_runtime.h>

typedef __attribute__((ext_vector_type(16))) __bf16 v16bf;
typedef __attribute__((ext_vector_type(8)))  float  v8f;

constexpr int Bn = 8;          // batch
constexpr int Cc = 512;        // in channels
constexpr int Nn = 6272;       // T*H*W = 8*28*28
constexpr int Mm = 1568;       // T*(H/2)*(W/2) = 8*14*14
constexpr int DIc = 256;       // dim_inner
constexpr int DOc = 512;       // dim_out

// ---------------- WMMA fragment helpers (16x16x32 bf16) ----------------
// A operand: memory layout [rows][K] row-major.  Per ISA: lane r=lane&15 holds
// row M=r; half h=lane>>4: elements 0..7 = K = h*8..h*8+7, elements 8..15 =
// K = 16+h*8 .. 16+h*8+7  -> two contiguous 16B chunks.
__device__ __forceinline__ v16bf ld_frag_a(const __bf16* p, int hf) {
  union { uint4 u[2]; v16bf v; } t;
  t.u[0] = *reinterpret_cast<const uint4*>(p + hf * 8);
  t.u[1] = *reinterpret_cast<const uint4*>(p + 16 + hf * 8);
  return t.v;
}
// B operand: memory layout [cols][K] (K contiguous). Per ISA: lane c=lane&15
// holds column N=c; half h: elements 0..7 = K = h*8..h*8+7, 8..15 = K = 16+h*8..16+h*8+7.
__device__ __forceinline__ v16bf ld_frag_b(const __bf16* p, int hf) {
  union { uint4 u[2]; v16bf v; } t;
  t.u[0] = *reinterpret_cast<const uint4*>(p + hf * 8);
  t.u[1] = *reinterpret_cast<const uint4*>(p + 16 + hf * 8);
  return t.v;
}
__device__ __forceinline__ v8f wmma_bf(v16bf a, v16bf b, v8f c) {
  return __builtin_amdgcn_wmma_f32_16x16x32_bf16(false, a, false, b, (short)0, c,
                                                 false, false);
}

// ---------------- small utility kernels ----------------
__global__ void cvt_bf16_kernel(const float* __restrict__ src,
                                __bf16* __restrict__ dst, int n) {
  int i = blockIdx.x * blockDim.x + threadIdx.x;
  if (i < n) dst[i] = (__bf16)src[i];
}

// x [B][C][N] f32 -> xT [B][N][C] bf16, LDS-tiled 32x32 transpose
__global__ __launch_bounds__(256) void transpose_x_kernel(
    const float* __restrict__ x, __bf16* __restrict__ xT) {
  __shared__ float tile[32][33];
  int b = blockIdx.z, c0 = blockIdx.y * 32, n0 = blockIdx.x * 32;
  const float* xp = x + ((size_t)b * Cc + c0) * Nn + n0;
  for (int i = threadIdx.y; i < 32; i += 8)
    tile[i][threadIdx.x] = xp[(size_t)i * Nn + threadIdx.x];
  __syncthreads();
  __bf16* xo = xT + ((size_t)b * Nn + n0) * Cc + c0;
  for (int i = threadIdx.y; i < 32; i += 8)
    xo[(size_t)i * Cc + threadIdx.x] = (__bf16)tile[threadIdx.x][i];
}

// 2x2 spatial max-pool done on channel-last layout: mpT [B][M][C] bf16
__global__ void maxpool_kernel(const __bf16* __restrict__ xT,
                               __bf16* __restrict__ mpT) {
  int idx = blockIdx.x * blockDim.x + threadIdx.x;
  if (idx >= Bn * Mm * Cc) return;
  int c = idx & (Cc - 1);
  int rem = idx >> 9;                 // /Cc
  int m = rem % Mm, b = rem / Mm;
  int t = m / 196, r = m % 196;
  int hh = r / 14, ww = r % 14;
  int n = t * 784 + (hh * 2) * 28 + ww * 2;
  const __bf16* p = xT + ((size_t)b * Nn + n) * Cc + c;
  float v0 = (float)p[0], v1 = (float)p[Cc];
  float v2 = (float)p[28 * Cc], v3 = (float)p[29 * Cc];
  mpT[idx] = (__bf16)fmaxf(fmaxf(v0, v1), fmaxf(v2, v3));
}

// ---------------- generic 16-row x (CT*16)-col WMMA GEMM ----------------
// D[row][col] = sum_k A[row][k]*Bm[col][k] + bias ; one wave per block.
// All shape/bias/output decisions are compile-time -> no conditional WMMA,
// no accumulator merging, branch-free epilogue.
template <int CT, bool BIASCOL, bool F32OUT>
__global__ __launch_bounds__(32) void gemm16_kernel(
    const __bf16* __restrict__ A, long long strideA, int lda,
    const __bf16* __restrict__ Bm, long long strideB, int ldb,
    const float* __restrict__ bias,
    __bf16* __restrict__ outB, float* __restrict__ outF,
    long long strideO, int ldo, int K) {
  int b = blockIdx.z;
  int row0 = blockIdx.x * 16, col0 = blockIdx.y * (CT * 16);
  int lane = threadIdx.x, l15 = lane & 15, hf = lane >> 4;
  A  += (size_t)b * strideA;
  Bm += (size_t)b * strideB;

  const v8f vz = {0.f, 0.f, 0.f, 0.f, 0.f, 0.f, 0.f, 0.f};
  v8f acc[CT];
#pragma unroll
  for (int ct = 0; ct < CT; ++ct) acc[ct] = vz;

  const __bf16* arow = A + (size_t)(row0 + l15) * lda;
  const __bf16* brow = Bm + (size_t)(col0 + l15) * ldb;
  for (int k0 = 0; k0 < K; k0 += 32) {
    v16bf af = ld_frag_a(arow + k0, hf);
#pragma unroll
    for (int ct = 0; ct < CT; ++ct) {
      v16bf bb = ld_frag_b(brow + (size_t)(ct * 16) * ldb + k0, hf);
      acc[ct] = wmma_bf(af, bb, acc[ct]);
    }
  }

  float brbias[8];
  if (!BIASCOL) {
#pragma unroll
    for (int r = 0; r < 8; ++r) brbias[r] = bias[row0 + r + 8 * hf];
  }
#pragma unroll
  for (int ct = 0; ct < CT; ++ct) {
    int col = col0 + ct * 16 + l15;
    float bcol = BIASCOL ? bias[col] : 0.f;
#pragma unroll
    for (int r = 0; r < 8; ++r) {
      int row = row0 + r + 8 * hf;
      float v = acc[ct][r] + (BIASCOL ? bcol : brbias[r]);
      if (F32OUT)
        outF[(size_t)b * strideO + (size_t)row * ldo + col] = v;
      else
        outB[(size_t)b * strideO + (size_t)row * ldo + col] = (__bf16)v;
    }
  }
}

// ---------------- fused attention: logits -> softmax -> P@g^T -------------
// One block (2 waves) handles 16 rows of N for one batch, sharing one LDS P
// buffer (bf16 [16][Mm], exp-unnormalized; 1/rowsum folded into epilogue).
// Wave split: stage1 interleaves m-tile PAIRS (two independent WMMA chains
// per wave for XDL ILP), stage2 gives each wave 8 softmax rows, stage3 gives
// each wave 8 of the 16 d-tiles (no cross-wave reduction).
// 2 waves / 50KB LDS -> ~12 resident waves per WGP.
__global__ __launch_bounds__(64) void attn_kernel(
    const __bf16* __restrict__ thetaT,  // [B][N][DI]
    const __bf16* __restrict__ phiT,    // [B][M][DI]
    const __bf16* __restrict__ gmat,    // [B][DI][M]
    __bf16* __restrict__ ttT) {         // [B][N][DI]
  __shared__ __bf16 P[16 * Mm];
  __shared__ float rowinv[16];
  int n0 = blockIdx.x * 16, b = blockIdx.y;
  int wv = threadIdx.x >> 5;            // wave id 0/1
  int lane = threadIdx.x & 31, l15 = lane & 15, hf = lane >> 4;
  const __bf16* th = thetaT + (size_t)b * Nn * DIc;
  const __bf16* ph = phiT   + (size_t)b * Mm * DIc;
  const __bf16* gg = gmat   + (size_t)b * DIc * Mm;

  // preload theta A-fragments for the 16 rows (K = 256 -> 8 fragments)
  v16bf ath[8];
  const __bf16* throw_ = th + (size_t)(n0 + l15) * DIc;
#pragma unroll
  for (int k = 0; k < 8; ++k) ath[k] = ld_frag_a(throw_ + k * 32, hf);

  const v8f vz = {0.f, 0.f, 0.f, 0.f, 0.f, 0.f, 0.f, 0.f};
  // stage 1: logits = (theta^T phi) * DI^-0.5, bf16 into LDS; each wave
  // takes a pair of adjacent m-tiles per iteration -> two independent
  // accumulator chains (uniform control per wave, EXEC stays full)
  for (int mtp = wv; mtp < Mm / 32; mtp += 2) {
    int mt0 = mtp * 2;
    v8f acc0 = vz, acc1 = vz;
    const __bf16* pb0 = ph + (size_t)(mt0 * 16 + l15) * DIc;
    const __bf16* pb1 = pb0 + (size_t)16 * DIc;
#pragma unroll
    for (int k = 0; k < 8; ++k) {
      acc0 = wmma_bf(ath[k], ld_frag_b(pb0 + k * 32, hf), acc0);
      acc1 = wmma_bf(ath[k], ld_frag_b(pb1 + k * 32, hf), acc1);
    }
#pragma unroll
    for (int r = 0; r < 8; ++r) {
      P[(r + 8 * hf) * Mm + mt0 * 16 + l15]      = (__bf16)(acc0[r] * 0.0625f);
      P[(r + 8 * hf) * Mm + mt0 * 16 + 16 + l15] = (__bf16)(acc1[r] * 0.0625f);
    }
  }
  __syncthreads();

  // stage 2: row softmax; wave wv owns rows wv*8..wv*8+7, 4 lanes per row
  {
    int row = wv * 8 + (lane & 7);
    int seg = lane >> 3;                       // 0..3
    __bf16* lr = P + row * Mm + seg * (Mm / 4);
    float mx = -3.0e38f;
    for (int j = 0; j < Mm / 4; ++j) mx = fmaxf(mx, (float)lr[j]);
    mx = fmaxf(mx, __shfl_xor(mx, 8));
    mx = fmaxf(mx, __shfl_xor(mx, 16));
    float s = 0.f;
    for (int j = 0; j < Mm / 4; ++j) {
      float e = __expf((float)lr[j] - mx);
      s += e;
      lr[j] = (__bf16)e;
    }
    s += __shfl_xor(s, 8);
    s += __shfl_xor(s, 16);
    if (seg == 0) rowinv[row] = 1.0f / s;
  }
  __syncthreads();

  // stage 3: ttT[n][d] = sum_m P[n][m] * g[d][m]; wave wv owns d-tiles
  // wv*8 .. wv*8+7 over the full K = Mm (8 independent chains)
  v8f accd[8];
#pragma unroll
  for (int dt = 0; dt < 8; ++dt) accd[dt] = vz;
  const __bf16* gbase = gg + (size_t)(wv * 8 * 16 + l15) * Mm;
  for (int mk = 0; mk < Mm / 32; ++mk) {
    v16bf af = ld_frag_a(P + l15 * Mm + mk * 32, hf);
#pragma unroll
    for (int dt = 0; dt < 8; ++dt) {
      v16bf bb = ld_frag_b(gbase + (size_t)(dt * 16) * Mm + mk * 32, hf);
      accd[dt] = wmma_bf(af, bb, accd[dt]);
    }
  }
  __bf16* op = ttT + (size_t)b * Nn * DIc + (size_t)(wv * 8 * 16);
#pragma unroll
  for (int r = 0; r < 8; ++r) {
    float inv = rowinv[r + 8 * hf];
    size_t rowoff = (size_t)(n0 + r + 8 * hf) * DIc;
#pragma unroll
    for (int dt = 0; dt < 8; ++dt)
      op[rowoff + dt * 16 + l15] = (__bf16)(accd[dt][r] * inv);
  }
}

// ---------------- BatchNorm ----------------
__global__ void bn_zero_kernel(float* s) {
  int i = blockIdx.x * blockDim.x + threadIdx.x;
  if (i < 2 * DOc) s[i] = 0.f;
}

// outT [B*N][DO] f32; per-channel sum & sumsq via block partials + atomics
__global__ __launch_bounds__(256) void bn_stats_kernel(
    const float* __restrict__ outT, float* __restrict__ sums) {
  int r0 = blockIdx.x * 392;      // 128 blocks * 392 rows = 50176 = B*N
  int c0 = threadIdx.x, c1 = threadIdx.x + 256;
  float s0 = 0.f, q0 = 0.f, s1 = 0.f, q1 = 0.f;
  for (int r = 0; r < 392; ++r) {
    const float* row = outT + (size_t)(r0 + r) * DOc;
    float a = row[c0], bb = row[c1];
    s0 += a; q0 += a * a; s1 += bb; q1 += bb * bb;
  }
  atomicAdd(&sums[c0], s0);       atomicAdd(&sums[DOc + c0], q0);
  atomicAdd(&sums[c1], s1);       atomicAdd(&sums[DOc + c1], q1);
}

__global__ void bn_params_kernel(const float* __restrict__ sums,
                                 const float* __restrict__ gamma,
                                 const float* __restrict__ beta,
                                 float* __restrict__ sc) {
  int o = blockIdx.x * blockDim.x + threadIdx.x;
  if (o >= DOc) return;
  float cnt = (float)Bn * (float)Nn;
  float mean = sums[o] / cnt;
  float var = sums[DOc + o] / cnt - mean * mean;
  float s = gamma[o] * rsqrtf(var + 1e-5f);
  sc[o] = s;
  sc[DOc + o] = beta[o] - mean * s;
}

// normalize + residual, write final [B][DO][N] layout (coalesced along N)
__global__ void bn_final_kernel(const float* __restrict__ outT,
                                const float* __restrict__ sc,
                                const float* __restrict__ x,
                                float* __restrict__ out) {
  size_t idx = (size_t)blockIdx.x * blockDim.x + threadIdx.x;
  if (idx >= (size_t)Bn * DOc * Nn) return;
  int n = (int)(idx % Nn);
  int o = (int)((idx / Nn) % DOc);
  int b = (int)(idx / ((size_t)Nn * DOc));
  float v = outT[((size_t)b * Nn + n) * DOc + o];
  out[idx] = v * sc[o] + sc[DOc + o] + x[idx];
}

// ---------------- launch ----------------
extern "C" void kernel_launch(void* const* d_in, const int* in_sizes, int n_in,
                              void* d_out, int out_size, void* d_ws,
                              size_t ws_size, hipStream_t stream) {
  const float* x       = (const float*)d_in[0];
  const float* theta_w = (const float*)d_in[1];
  const float* theta_b = (const float*)d_in[2];
  const float* phi_w   = (const float*)d_in[3];
  const float* phi_b   = (const float*)d_in[4];
  const float* g_w     = (const float*)d_in[5];
  const float* g_b     = (const float*)d_in[6];
  const float* out_w   = (const float*)d_in[7];
  const float* out_b   = (const float*)d_in[8];
  const float* gamma   = (const float*)d_in[9];
  const float* beta    = (const float*)d_in[10];
  float* out = (float*)d_out;

  char* p = (char*)d_ws;
  auto take = [&](size_t bytes) -> char* {
    char* r = p;
    p += (bytes + 255) & ~(size_t)255;
    return r;
  };
  __bf16* xT     = (__bf16*)take((size_t)Bn * Nn * Cc * 2);   // 51.4 MB
  __bf16* mpT    = (__bf16*)take((size_t)Bn * Mm * Cc * 2);   // 12.8 MB
  __bf16* thW    = (__bf16*)take((size_t)DIc * Cc * 2);
  __bf16* phW    = (__bf16*)take((size_t)DIc * Cc * 2);
  __bf16* gW     = (__bf16*)take((size_t)DIc * Cc * 2);
  __bf16* outW   = (__bf16*)take((size_t)DOc * DIc * 2);
  __bf16* thetaT = (__bf16*)take((size_t)Bn * Nn * DIc * 2);  // 25.7 MB
  __bf16* phiT   = (__bf16*)take((size_t)Bn * Mm * DIc * 2);  // 6.4 MB
  __bf16* gB     = (__bf16*)take((size_t)Bn * DIc * Mm * 2);  // 6.4 MB
  __bf16* ttT    = (__bf16*)take((size_t)Bn * Nn * DIc * 2);  // 25.7 MB
  float*  outT   = (float*)take((size_t)Bn * Nn * DOc * 4);   // 102.8 MB
  float*  bnS    = (float*)take((size_t)4 * DOc * 4);         // sums + scale/shift

  // weights -> bf16
  cvt_bf16_kernel<<<512, 256, 0, stream>>>(theta_w, thW, DIc * Cc);
  cvt_bf16_kernel<<<512, 256, 0, stream>>>(phi_w, phW, DIc * Cc);
  cvt_bf16_kernel<<<512, 256, 0, stream>>>(g_w, gW, DIc * Cc);
  cvt_bf16_kernel<<<512, 256, 0, stream>>>(out_w, outW, DOc * DIc);

  // x -> channel-last bf16; max-pool
  transpose_x_kernel<<<dim3(Nn / 32, Cc / 32, Bn), dim3(32, 8), 0, stream>>>(x, xT);
  maxpool_kernel<<<(Bn * Mm * Cc + 255) / 256, 256, 0, stream>>>(xT, mpT);

  // projections (grids exact in cols -> no runtime guards)
  // thetaT[N][DI] = xT x theta_w   (4 col-tiles * 4 groups = 256)
  gemm16_kernel<4, true, false><<<dim3(Nn / 16, DIc / 64, Bn), 32, 0, stream>>>(
      xT, (long long)Nn * Cc, Cc, thW, 0, Cc, theta_b,
      thetaT, nullptr, (long long)Nn * DIc, DIc, Cc);
  // phiT[M][DI] = mpT x phi_w
  gemm16_kernel<4, true, false><<<dim3(Mm / 16, DIc / 64, Bn), 32, 0, stream>>>(
      mpT, (long long)Mm * Cc, Cc, phW, 0, Cc, phi_b,
      phiT, nullptr, (long long)Mm * DIc, DIc, Cc);
  // g[DI][M] = g_w x mpT   (2 col-tiles * 49 groups = 1568)
  gemm16_kernel<2, false, false><<<dim3(DIc / 16, Mm / 32, Bn), 32, 0, stream>>>(
      gW, 0, Cc, mpT, (long long)Mm * Cc, Cc, g_b,
      gB, nullptr, (long long)DIc * Mm, Mm, Cc);

  // fused attention (2 waves / block)
  attn_kernel<<<dim3(Nn / 16, Bn), 64, 0, stream>>>(thetaT, phiT, gB, ttT);

  // output projection (f32 out for BN)  (4 col-tiles * 8 groups = 512)
  gemm16_kernel<4, true, true><<<dim3(Nn / 16, DOc / 64, Bn), 32, 0, stream>>>(
      ttT, (long long)Nn * DIc, DIc, outW, 0, DIc, out_b,
      nullptr, outT, (long long)Nn * DOc, DOc, DIc);

  // BatchNorm + residual
  bn_zero_kernel<<<4, 256, 0, stream>>>(bnS);
  bn_stats_kernel<<<128, 256, 0, stream>>>(outT, bnS);
  bn_params_kernel<<<2, 256, 0, stream>>>(bnS, gamma, beta, bnS + 2 * DOc);
  bn_final_kernel<<<(int)(((size_t)Bn * DOc * Nn + 255) / 256), 256, 0, stream>>>(
      outT, bnS + 2 * DOc, x, out);
}
